// SemanticSubspaceAttentionV2_11871289606896
// MI455X (gfx1250) — matrix operand order repaired and checked by reading, not compiled, not run on hardware
//
#include <hip/hip_runtime.h>
#include <hip/hip_bf16.h>
#include <cstdint>

typedef _Float16 v16h __attribute__((ext_vector_type(16)));
typedef _Float16 v8h  __attribute__((ext_vector_type(8)));
typedef float    v8f  __attribute__((ext_vector_type(8)));

#define B_ 2
#define N_ 1024
#define M_ 1024
#define S_ 4
#define D_ 256
#define H_ 8
#define C_ 32

// ---------------------------------------------------------------------------
// Kernel 1: f32 -> f16 conversion of inputs; W transposed so that the WMMA
// B-fragment (lane = N column; per K row 16 lanes read 16 contiguous f16 = 32B) loads coalesced.
// ---------------------------------------------------------------------------
__global__ void __launch_bounds__(256)
cvt_kernel(const float* __restrict__ xq, const float* __restrict__ xk,
           const float* __restrict__ xv,
           const float* __restrict__ Wq, const float* __restrict__ Wk,
           const float* __restrict__ Wv,
           _Float16* __restrict__ xqh, _Float16* __restrict__ xkh,
           _Float16* __restrict__ xvh,
           _Float16* __restrict__ wtq, _Float16* __restrict__ wtk,
           _Float16* __restrict__ wtv)
{
    int i = blockIdx.x * 256 + threadIdx.x;
    if (i < B_ * N_ * D_) {
        xqh[i] = (_Float16)xq[i];
        xkh[i] = (_Float16)xk[i];
        xvh[i] = (_Float16)xv[i];
    }
    if (i < D_ * D_) {
        int d = i >> 8, e = i & 255;          // W[d][e] -> Wt[e][d]
        int o = e * D_ + d;
        wtq[o] = (_Float16)Wq[i];
        wtk[o] = (_Float16)Wk[i];
        wtv[o] = (_Float16)Wv[i];
    }
}

// Build the 16-bit A fragment (16x32, MxK) from a row-major f16 row pointer.
// Lanes 0-15: row M=lane, K = {0..7, 16..23}; lanes 16-31: same row, K = {8..15, 24..31}.
__device__ inline v16h load_a_frag_f16(const _Float16* __restrict__ row, int k0, int lane)
{
    int koff = (lane & 16) ? 8 : 0;
    v8h lo = *(const v8h*)(row + k0 + koff);
    v8h hi = *(const v8h*)(row + k0 + 16 + koff);
    v16h a;
#pragma unroll
    for (int i = 0; i < 8; ++i) { a[i] = lo[i]; a[i + 8] = hi[i]; }
    return a;
}

// ---------------------------------------------------------------------------
// Kernel 2: QKV projection GEMMs: out[n,d] = sum_e X[n,e] * W[d,e] + bias[d]
// (A = X row-major, B = Wt row-major). Epilogue scatters to head-major f16:
//   q  -> [B,H,N,C]   k -> kt [B,H,C,M]   v -> [B,H,M,C]
// ---------------------------------------------------------------------------
__global__ void __launch_bounds__(128)
proj_wmma_kernel(const _Float16* __restrict__ xh,   // 3 matrices, each B_*N_*D_
                 const _Float16* __restrict__ wth,  // 3 matrices, each D_*D_
                 const float* __restrict__ bq, const float* __restrict__ bk,
                 const float* __restrict__ bv,
                 _Float16* __restrict__ qh, _Float16* __restrict__ kth,
                 _Float16* __restrict__ vh)
{
    int lane = threadIdx.x & 31;
    int wave = threadIdx.x >> 5;
    int tile = blockIdx.x * 4 + wave;          // 0..6143
    int mat  = tile / 2048;                    // 0=q 1=k 2=v
    int rem  = tile % 2048;
    int rt   = rem & 127;                      // token-row tile (2048/16)
    int ct   = rem >> 7;                       // d-col tile (256/16)
    int n0 = rt * 16, d0 = ct * 16;

    const _Float16* X    = xh  + (size_t)mat * (B_ * N_ * D_);
    const _Float16* Wt   = wth + (size_t)mat * (D_ * D_);
    const float*    bias = (mat == 0) ? bq : (mat == 1 ? bk : bv);

    const _Float16* arow = X + (size_t)(n0 + (lane & 15)) * D_;
    v8f acc = {};
#pragma unroll
    for (int kt = 0; kt < 8; ++kt) {
        int k0 = kt * 32;
        v16h a = load_a_frag_f16(arow, k0, lane);
        // B fragment: lane holds column N=d0+lane%16 of Wt, K in same order as A
        v16h bfr;
        int bko = (lane & 16) ? 8 : 0;
#pragma unroll
        for (int i = 0; i < 8; ++i) {
            bfr[i]     = Wt[(size_t)(k0 + bko + i) * D_ + d0 + (lane & 15)];
            bfr[i + 8] = Wt[(size_t)(k0 + 16 + bko + i) * D_ + d0 + (lane & 15)];
        }
        acc = __builtin_amdgcn_wmma_f32_16x16x32_f16(false, a, false, bfr,
                                                     (short)0, acc, false, false);
    }
    int rbase = (lane & 16) ? 8 : 0;
    int col   = d0 + (lane & 15);
    int h = col >> 5, c = col & 31;
#pragma unroll
    for (int e = 0; e < 8; ++e) {
        int rowg = n0 + e + rbase;             // 0..2047  (flattened b*len + t)
        int b = rowg >> 10, t = rowg & 1023;
        _Float16 hv = (_Float16)(acc[e] + bias[col]);
        if (mat == 0)      qh [(((size_t)b * H_ + h) * N_ + t) * C_ + c] = hv;
        else if (mat == 1) kth[(((size_t)b * H_ + h) * C_ + c) * M_ + t] = hv;
        else               vh [(((size_t)b * H_ + h) * M_ + t) * C_ + c] = hv;
    }
}

// ---------------------------------------------------------------------------
// Kernel 3: fused QK^T (one WMMA per 16x16 tile, K=C=32 exactly) + per-s
// scaling/masking + softmax over M, streaming normalized attn to d_out once.
// One workgroup (4 waves) owns a 16-row x 1024-col strip for one (b,h).
// ---------------------------------------------------------------------------
__global__ void __launch_bounds__(128)
scores_softmax_kernel(const _Float16* __restrict__ qh,
                      const _Float16* __restrict__ kth,
                      const float* __restrict__ qsub,   // [B,S,N]
                      const float* __restrict__ ksub,   // [B,S,M]
                      const float* __restrict__ keyw,   // [B,M]
                      const unsigned char* __restrict__ keym,   // [B,M] bool
                      const float* __restrict__ factors,        // [B,N,M]
                      const unsigned char* __restrict__ attm,   // [B,N,M] bool
                      float* __restrict__ attn_out)             // [B,S,H,N,M]
{
    __shared__ __align__(16) float lds1[16][M_];   // raw QK^T strip
    __shared__ __align__(16) float lds2[16][M_];   // per-s scaled / exp strip

    int lane = threadIdx.x & 31, wave = threadIdx.x >> 5;
    int bid = blockIdx.x;                 // 0..1023
    int nt = bid & 63, h = (bid >> 6) & 7, b = bid >> 9;
    int n0 = nt * 16;

    // A fragment: q rows, K = C = 32 (single k-step)
    const _Float16* arow =
        qh + (((size_t)b * H_ + h) * N_ + n0 + (lane & 15)) * C_;
    v16h a = load_a_frag_f16(arow, 0, lane);

    // B fragment base: lane holds column m=m0+lane%16 of k^T, K=c in same order as A
    const _Float16* krow = kth + (((size_t)b * H_ + h) * C_) * M_;
    int rbase = (lane & 16) ? 8 : 0;

    for (int mt = wave; mt < 64; mt += 4) {
        int m0 = mt * 16;
        v16h bfr;
#pragma unroll
        for (int i = 0; i < 8; ++i) {
            bfr[i]     = krow[(size_t)(rbase + i) * M_ + m0 + (lane & 15)];
            bfr[i + 8] = krow[(size_t)(16 + rbase + i) * M_ + m0 + (lane & 15)];
        }
        v8f acc = {};
        acc = __builtin_amdgcn_wmma_f32_16x16x32_f16(false, a, false, bfr,
                                                     (short)0, acc, false, false);
#pragma unroll
        for (int e = 0; e < 8; ++e)
            lds1[e + rbase][m0 + (lane & 15)] = acc[e];
    }
    __syncthreads();

    const float inv_sqrtC = 0.17677669529663687f;   // 1/sqrt(32)
    int row = threadIdx.x >> 3;    // 0..15
    int seg = threadIdx.x & 7;     // 8 threads per row, 128 cols each
    int n = n0 + row;

    const float*         frow  = factors + ((size_t)b * N_ + n) * M_;
    const unsigned char* amrow = attm    + ((size_t)b * N_ + n) * M_;
    const float*         kwrow = keyw + (size_t)b * M_;
    const unsigned char* kmrow = keym + (size_t)b * M_;

    for (int s = 0; s < S_; ++s) {
        float qs = qsub[((size_t)b * S_ + s) * N_ + n];
        const float* ksr = ksub + ((size_t)b * S_ + s) * M_;

        float maxv = -__builtin_inff();
        for (int j = 0; j < 32; ++j) {
            int m = seg * 128 + j * 4;
            float4 base = *(const float4*)&lds1[row][m];
            float4 f    = *(const float4*)(frow + m);
            float4 kw   = *(const float4*)(kwrow + m);
            float4 ks   = *(const float4*)(ksr + m);
            unsigned msk = *(const unsigned*)(kmrow + m) |
                           *(const unsigned*)(amrow + m);
            float4 v;
            v.x = (msk & 0x000000ffu) ? -__builtin_inff()
                                      : base.x * inv_sqrtC * f.x * kw.x * qs * ks.x;
            v.y = (msk & 0x0000ff00u) ? -__builtin_inff()
                                      : base.y * inv_sqrtC * f.y * kw.y * qs * ks.y;
            v.z = (msk & 0x00ff0000u) ? -__builtin_inff()
                                      : base.z * inv_sqrtC * f.z * kw.z * qs * ks.z;
            v.w = (msk & 0xff000000u) ? -__builtin_inff()
                                      : base.w * inv_sqrtC * f.w * kw.w * qs * ks.w;
            *(float4*)&lds2[row][m] = v;
            maxv = fmaxf(maxv, fmaxf(fmaxf(v.x, v.y), fmaxf(v.z, v.w)));
        }
        maxv = fmaxf(maxv, __shfl_xor(maxv, 1, 8));
        maxv = fmaxf(maxv, __shfl_xor(maxv, 2, 8));
        maxv = fmaxf(maxv, __shfl_xor(maxv, 4, 8));

        float sum = 0.f;
        for (int j = 0; j < 32; ++j) {
            int m = seg * 128 + j * 4;
            float4 v = *(const float4*)&lds2[row][m];
            v.x = __expf(v.x - maxv);
            v.y = __expf(v.y - maxv);
            v.z = __expf(v.z - maxv);
            v.w = __expf(v.w - maxv);
            *(float4*)&lds2[row][m] = v;
            sum += (v.x + v.y) + (v.z + v.w);
        }
        sum += __shfl_xor(sum, 1, 8);
        sum += __shfl_xor(sum, 2, 8);
        sum += __shfl_xor(sum, 4, 8);
        float inv = 1.0f / sum;

        float* orow = attn_out +
            ((((size_t)(b * S_ + s) * H_ + h) * N_ + n) * M_);
        for (int j = 0; j < 32; ++j) {
            int m = seg * 128 + j * 4;
            float4 v = *(const float4*)&lds2[row][m];
            v.x *= inv; v.y *= inv; v.z *= inv; v.w *= inv;
            *(float4*)(orow + m) = v;
        }
    }
}

// ---------------------------------------------------------------------------
// Kernel 4: hidden[b,s,n, h*32+c] = sum_m attn[b,s,h,n,m] * v[b,h,m,c]
// One wave per 16-row x 32-col (full head) output tile; K=M in 32-chunks,
// 2 WMMAs per k-step; attn converted f32->f16 on load.
// ---------------------------------------------------------------------------
__global__ void __launch_bounds__(128)
av_wmma_kernel(const float* __restrict__ attn, const _Float16* __restrict__ vh,
               float* __restrict__ hidden)
{
    int lane = threadIdx.x & 31, wave = threadIdx.x >> 5;
    int tile = blockIdx.x * 4 + wave;          // 0..4095
    int nt = tile & 63, h = (tile >> 6) & 7, s = (tile >> 9) & 3, b = tile >> 11;
    int n0 = nt * 16;

    const float* arow = attn +
        ((((size_t)(b * S_ + s) * H_ + h) * N_ + n0 + (lane & 15)) * M_);
    int koff = (lane & 16) ? 8 : 0;
    const _Float16* vbase = vh + ((size_t)b * H_ + h) * (size_t)M_ * C_;

    v8f acc0 = {}, acc1 = {};
    for (int mt = 0; mt < 32; ++mt) {
        int m0 = mt * 32;
        __builtin_prefetch(arow + m0 + 64, 0, 1);
        v16h a;
#pragma unroll
        for (int i = 0; i < 8; ++i) {
            a[i]     = (_Float16)arow[m0 + koff + i];
            a[i + 8] = (_Float16)arow[m0 + 16 + koff + i];
        }
        // B fragment: lane holds column c=lane%16 (b1: +16) of v, K=m in same order as A
        const _Float16* brow = vbase + (size_t)m0 * C_ + (lane & 15);
        v16h b0, b1;
#pragma unroll
        for (int i = 0; i < 8; ++i) {
            b0[i]     = brow[(size_t)(koff + i) * C_];
            b0[i + 8] = brow[(size_t)(16 + koff + i) * C_];
            b1[i]     = brow[(size_t)(koff + i) * C_ + 16];
            b1[i + 8] = brow[(size_t)(16 + koff + i) * C_ + 16];
        }
        acc0 = __builtin_amdgcn_wmma_f32_16x16x32_f16(false, a, false, b0,
                                                      (short)0, acc0, false, false);
        acc1 = __builtin_amdgcn_wmma_f32_16x16x32_f16(false, a, false, b1,
                                                      (short)0, acc1, false, false);
    }
    int rbase = (lane & 16) ? 8 : 0;
    int c0 = h * 32 + (lane & 15);
#pragma unroll
    for (int e = 0; e < 8; ++e) {
        int n = n0 + e + rbase;
        float* hrow = hidden + (((size_t)(b * S_ + s) * N_ + n) * D_);
        hrow[c0]      = acc0[e];
        hrow[c0 + 16] = acc1[e];
    }
}

// ---------------------------------------------------------------------------
extern "C" void kernel_launch(void* const* d_in, const int* in_sizes, int n_in,
                              void* d_out, int out_size, void* d_ws, size_t ws_size,
                              hipStream_t stream)
{
    const float* xq      = (const float*)d_in[0];
    const float* xk      = (const float*)d_in[1];
    const float* xv      = (const float*)d_in[2];
    const float* qsub    = (const float*)d_in[3];
    const float* ksub    = (const float*)d_in[4];
    const float* keyw    = (const float*)d_in[5];
    const unsigned char* keym = (const unsigned char*)d_in[6];
    const float* factors = (const float*)d_in[7];
    const unsigned char* attm = (const unsigned char*)d_in[8];
    const float* Wq = (const float*)d_in[9];
    const float* bq = (const float*)d_in[10];
    const float* Wk = (const float*)d_in[11];
    const float* bk = (const float*)d_in[12];
    const float* Wv = (const float*)d_in[13];
    const float* bv = (const float*)d_in[14];

    // Workspace layout (f16 elements): ~6.7 MB total
    _Float16* ws  = (_Float16*)d_ws;
    _Float16* xh  = ws;                                 // 3 * 524288 (xq,xk,xv f16)
    _Float16* wth = ws + (size_t)3 * 524288;            // 3 * 65536  (Wt f16)
    _Float16* qh  = wth + (size_t)3 * 65536;            // [B,H,N,C]
    _Float16* kth = qh  + 524288;                       // [B,H,C,M]
    _Float16* vh  = kth + 524288;                       // [B,H,M,C]

    float* hidden = (float*)d_out;                      // [B,S,N,D]
    float* attn   = hidden + (size_t)B_ * S_ * N_ * D_; // [B,S,H,N,M]

    cvt_kernel<<<2048, 256, 0, stream>>>(xq, xk, xv, Wq, Wk, Wv,
                                         xh, xh + 524288, xh + 2 * 524288,
                                         wth, wth + 65536, wth + 2 * 65536);
    proj_wmma_kernel<<<1536, 128, 0, stream>>>(xh, wth, bq, bk, bv, qh, kth, vh);
    scores_softmax_kernel<<<1024, 128, 0, stream>>>(qh, kth, qsub, ksub, keyw,
                                                    keym, factors, attm, attn);
    av_wmma_kernel<<<1024, 128, 0, stream>>>(attn, vh, hidden);
}
